// GCNNet_31971736551526
// MI455X (gfx1250) — compile-verified, matrix operand order checked by reading, not run on hardware
//
#include <hip/hip_runtime.h>
#include <cstdint>
#include <cstddef>

// ---------------------------------------------------------------------------
// Types for CDNA5 WMMA (wave32): D = A(16x32 bf16) x B(32x16 bf16) + C(16x16 f32)
// ---------------------------------------------------------------------------
typedef __bf16 bf16_t;
typedef __attribute__((ext_vector_type(16))) __bf16 bf16x16;
typedef __attribute__((ext_vector_type(8)))  float  f32x8;

#define BN_EPS 1e-5f

// ---------------------------------------------------------------------------
// WMMA fragment swizzle (ISA 7.12.2, 16-bit A 16x32 / symmetric B 32x16):
//   lanes 0-15  : row/col = lane,    elems 0..7 = K 0..7,  elems 8..15 = K 16..23
//   lanes 16-31 : row/col = lane-16, elems 0..7 = K 8..15, elems 8..15 = K 24..31
// Buffer layout: [tile][kchunk][lane][elem] with 16 contiguous bf16 per lane.
// ---------------------------------------------------------------------------
__device__ __forceinline__ size_t frag_off(int tile, int c, int nch, int l15, int krem) {
    int lane = l15 + ((krem & 8) << 1);           // +16 if krem in {8..15, 24..31}
    int elem = (krem & 7) + ((krem & 16) >> 1);   // +8  if krem in {16..31}
    return (((size_t)tile * nch + c) * 32 + lane) * 16 + elem;
}

// ---------------------------------------------------------------------------
// Degree computation (weighted in-degree with self-loop weight 1.0)
// ---------------------------------------------------------------------------
__global__ void deg_init_kernel(float* __restrict__ deg, int n) {
    int i = blockIdx.x * blockDim.x + threadIdx.x;
    if (i < n) deg[i] = 1.0f;  // self-loop contributes weight 1
}

__global__ void deg_acc_kernel(const int* __restrict__ dst,
                               const float* __restrict__ w,
                               float* __restrict__ deg, int E) {
    int e = blockIdx.x * blockDim.x + threadIdx.x;
    if (e < E) atomicAdd(&deg[dst[e]], w[e]);
}

__global__ void deg_inv_kernel(const float* __restrict__ deg,
                               float* __restrict__ dinv, int n) {
    int i = blockIdx.x * blockDim.x + threadIdx.x;
    if (i < n) dinv[i] = rsqrtf(fmaxf(deg[i], 1e-12f));
}

// ---------------------------------------------------------------------------
// Weight conversion f32 [K,M] -> bf16 fragment-swizzled [K,Mpad], zero padded
// ---------------------------------------------------------------------------
__global__ void conv_w_swz_kernel(const float* __restrict__ W, bf16_t* __restrict__ out,
                                  int K, int M, int Mpad) {
    int idx = blockIdx.x * blockDim.x + threadIdx.x;
    if (idx >= K * Mpad) return;
    int k = idx / Mpad;
    int m = idx - k * Mpad;
    float v = (m < M) ? W[k * M + m] : 0.0f;
    out[frag_off(m >> 4, k >> 5, K >> 5, m & 15, k & 31)] = (bf16_t)v;
}

// ---------------------------------------------------------------------------
// BN (+optional bias, +optional ReLU) epilogue; writes bf16 activations in
// WMMA fragment order for the next GEMM. F power of two (128/64/32).
// ---------------------------------------------------------------------------
__global__ void bn_act_swz_kernel(const float* __restrict__ in,
                                  const float* __restrict__ bias,
                                  const float* __restrict__ g,
                                  const float* __restrict__ beta,
                                  const float* __restrict__ mu,
                                  const float* __restrict__ var,
                                  bf16_t* __restrict__ out,
                                  int total, int Fmask, int Fshift, int relu) {
    int idx = blockIdx.x * blockDim.x + threadIdx.x;
    if (idx >= total) return;
    int f = idx & Fmask;
    int row = idx >> Fshift;
    float v = in[idx];
    if (bias) v += bias[f];
    v = (v - mu[f]) * rsqrtf(var[f] + BN_EPS) * g[f] + beta[f];
    if (relu) v = fmaxf(v, 0.0f);
    int nch = (Fmask + 1) >> 5;
    out[frag_off(row >> 4, f >> 5, nch, row & 15, f & 31)] = (bf16_t)v;
}

// ---------------------------------------------------------------------------
// Self-loop contribution: agg[i,:] = H[i,:] * dinv[i]^2  (norm of self edge)
// ---------------------------------------------------------------------------
__global__ void agg_init_kernel(const float* __restrict__ H,
                                const float* __restrict__ dinv,
                                float* __restrict__ out, int total, int Fshift) {
    int idx = blockIdx.x * blockDim.x + threadIdx.x;
    if (idx >= total) return;
    int i = idx >> Fshift;
    float di = dinv[i];
    out[idx] = H[idx] * di * di;
}

// ---------------------------------------------------------------------------
// Edge scatter: out[dst,:] += H[src,:] * (dinv[src]*w*dinv[dst])
// One thread handles one (edge, group-of-4-features). gshift = log2(F/4).
// ---------------------------------------------------------------------------
__global__ void edge_scatter_kernel(const int* __restrict__ src,
                                    const int* __restrict__ dst,
                                    const float* __restrict__ w,
                                    const float* __restrict__ dinv,
                                    const float* __restrict__ H,
                                    float* __restrict__ out,
                                    long long total, int F, int gshift) {
    long long gid = (long long)blockIdx.x * blockDim.x + threadIdx.x;
    if (gid >= total) return;
    int e  = (int)(gid >> gshift);
    int fg = (int)(gid & ((1 << gshift) - 1));
    int s = src[e];
    int d = dst[e];
    float nrm = dinv[s] * w[e] * dinv[d];
    const float4 h4 = *(const float4*)(H + (size_t)s * F + fg * 4);
    float* o = out + (size_t)d * F + fg * 4;
    atomicAdd(o + 0, h4.x * nrm);
    atomicAdd(o + 1, h4.y * nrm);
    atomicAdd(o + 2, h4.z * nrm);
    atomicAdd(o + 3, h4.w * nrm);
}

// ---------------------------------------------------------------------------
// WMMA GEMM: C[N,M] = A[N,K](bf16, fragment-swizzled)
//                   * B[K,Mpad](bf16, fragment-swizzled) (+ bias[M]).
// One wave owns a full 16-row stripe: loads its A fragments once (NCH * 32B),
// then iterates all column tiles with fully unrolled WMMA chains.
// ---------------------------------------------------------------------------
template <int K, int M, int Mpad>
__global__ void gemm_wmma_kernel(const bf16_t* __restrict__ A,
                                 const bf16_t* __restrict__ B,
                                 const float* __restrict__ bias,
                                 float* __restrict__ C, int Nrows) {
    constexpr int NCH = K / 32;    // K-chunks of 32
    constexpr int CT  = Mpad / 16; // column tiles

    int rowTile = blockIdx.x * (blockDim.x >> 5) + (threadIdx.x >> 5);
    int lane = threadIdx.x & 31;
    int row0 = rowTile << 4;
    if (row0 >= Nrows) return;     // wave-uniform exit (EXEC stays full)

    // ---- A fragments for this stripe: NCH contiguous 32B loads ----
    bf16x16 af[NCH];
    const bf16_t* ap = A + ((size_t)rowTile * NCH * 32 + lane) * 16;
#pragma unroll
    for (int c = 0; c < NCH; ++c)
        af[c] = *(const bf16x16*)(ap + (size_t)c * 32 * 16);

    int l15 = lane & 15;
    int halfSel = lane >> 4;
    bool fullRows = (row0 + 16) <= Nrows;

#pragma unroll
    for (int ct = 0; ct < CT; ++ct) {
        const bf16_t* bp = B + ((size_t)ct * NCH * 32 + lane) * 16;
        f32x8 acc = {};
#pragma unroll
        for (int c = 0; c < NCH; ++c) {
            bf16x16 bf = *(const bf16x16*)(bp + (size_t)c * 32 * 16);
            acc = __builtin_amdgcn_wmma_f32_16x16x32_bf16(
                /*neg_a=*/false, af[c], /*neg_b=*/false, bf,
                /*c_mod=*/(short)0, acc, /*reuse_a=*/false, /*reuse_b=*/false);
        }
        // ---- store 16x16 tile: vgpr v -> row (row0 + v + 8*halfSel), col ----
        int col = ct * 16 + l15;
        int r0 = row0 + 8 * halfSel;
        float badd = 0.0f;
        if (bias) badd = (M == Mpad || col < M) ? bias[col] : 0.0f;
        float* cp = C + (size_t)r0 * M + col;
        if ((M == Mpad) && fullRows) {
#pragma unroll
            for (int v = 0; v < 8; ++v) cp[(size_t)v * M] = acc[v] + badd;
        } else {
#pragma unroll
            for (int v = 0; v < 8; ++v)
                if (col < M && (r0 + v) < Nrows) cp[(size_t)v * M] = acc[v] + badd;
        }
    }
}

// ---------------------------------------------------------------------------
// Host-side orchestration
// ---------------------------------------------------------------------------
static inline int cdiv(long long a, int b) { return (int)((a + b - 1) / b); }

extern "C" void kernel_launch(void* const* d_in, const int* in_sizes, int n_in,
                              void* d_out, int out_size, void* d_ws, size_t ws_size,
                              hipStream_t stream) {
    const float* x   = (const float*)d_in[0];
    const int*   ei  = (const int*)d_in[1];
    const float* ew  = (const float*)d_in[2];
    const float* W1  = (const float*)d_in[3];   const float* c1b = (const float*)d_in[4];
    const float* W2  = (const float*)d_in[5];   const float* c2b = (const float*)d_in[6];
    const float* W3  = (const float*)d_in[7];   const float* c3b = (const float*)d_in[8];
    const float* W4  = (const float*)d_in[9];   const float* c4b = (const float*)d_in[10];
    const float* lw1 = (const float*)d_in[11];  const float* lb1 = (const float*)d_in[12];
    const float* lw2 = (const float*)d_in[13];  const float* lb2 = (const float*)d_in[14];
    // bn[j] = {g, beta, mu, var} for j = 0..5
    const float* bn[6][4];
    for (int j = 0; j < 6; ++j)
        for (int k = 0; k < 4; ++k)
            bn[j][k] = (const float*)d_in[15 + 4 * j + k];

    const int N = in_sizes[0] / 128;
    const int E = in_sizes[2];
    const int C = in_sizes[14];          // 40
    const int Cpad = 48;                 // next multiple of 16

    const int* src = ei;
    const int* dst = ei + E;

    // ---- workspace carve-up (256B aligned chunks) ----
    char* p = (char*)d_ws;
    auto carve = [&](size_t bytes) -> char* {
        char* r = p;
        p += (bytes + 255) & ~(size_t)255;
        return r;
    };
    float*  deg  = (float*)carve((size_t)N * 4);
    float*  dinv = (float*)carve((size_t)N * 4);
    float*  H    = (float*)carve((size_t)N * 128 * 4);   // GEMM output (row major)
    float*  AGG  = (float*)carve((size_t)N * 128 * 4);   // aggregation output
    bf16_t* Xb   = (bf16_t*)carve((size_t)N * 128 * 2);  // bf16 activations (swizzled)
    bf16_t* W1p  = (bf16_t*)carve(128 * 128 * 2);
    bf16_t* W2p  = (bf16_t*)carve(128 * 128 * 2);
    bf16_t* W3p  = (bf16_t*)carve(128 * 64 * 2);
    bf16_t* W4p  = (bf16_t*)carve(64 * 32 * 2);
    bf16_t* L1p  = (bf16_t*)carve(32 * 64 * 2);
    bf16_t* L2p  = (bf16_t*)carve((size_t)64 * Cpad * 2);

    const int TB = 256;
    const int rowTiles = (N + 15) >> 4;
    const int gemmBlocks = cdiv(rowTiles, TB >> 5);

    // ---- degrees ----
    deg_init_kernel<<<cdiv(N, TB), TB, 0, stream>>>(deg, N);
    deg_acc_kernel<<<cdiv(E, TB), TB, 0, stream>>>(dst, ew, deg, E);
    deg_inv_kernel<<<cdiv(N, TB), TB, 0, stream>>>(deg, dinv, N);

    // ---- weights -> bf16 fragment-swizzled (padded) ----
    conv_w_swz_kernel<<<cdiv(128 * 128, TB), TB, 0, stream>>>(W1, W1p, 128, 128, 128);
    conv_w_swz_kernel<<<cdiv(128 * 128, TB), TB, 0, stream>>>(W2, W2p, 128, 128, 128);
    conv_w_swz_kernel<<<cdiv(128 * 64, TB), TB, 0, stream>>>(W3, W3p, 128, 64, 64);
    conv_w_swz_kernel<<<cdiv(64 * 32, TB), TB, 0, stream>>>(W4, W4p, 64, 32, 32);
    conv_w_swz_kernel<<<cdiv(32 * 64, TB), TB, 0, stream>>>(lw1, L1p, 32, 64, 64);
    conv_w_swz_kernel<<<cdiv(64 * Cpad, TB), TB, 0, stream>>>(lw2, L2p, 64, C, Cpad);

    auto aggregate = [&](const float* Hin, float* Out, int F, int Fshift) {
        int total = N * F;
        agg_init_kernel<<<cdiv(total, TB), TB, 0, stream>>>(Hin, dinv, Out, total, Fshift);
        int gshift = Fshift - 2;                         // groups of 4 features
        long long etot = (long long)E << gshift;
        edge_scatter_kernel<<<cdiv(etot, TB), TB, 0, stream>>>(
            src, dst, ew, dinv, Hin, Out, etot, F, gshift);
    };

    auto bn_act = [&](const float* in, const float* bias, int bnIdx,
                      bf16_t* out, int F, int Fshift, int relu) {
        int total = N * F;
        bn_act_swz_kernel<<<cdiv(total, TB), TB, 0, stream>>>(
            in, bias, bn[bnIdx][0], bn[bnIdx][1], bn[bnIdx][2], bn[bnIdx][3],
            out, total, F - 1, Fshift, relu);
    };

    // ---- input BN1 -> bf16 (swizzled) ----
    bn_act(x, nullptr, 0, Xb, 128, 7, /*relu=*/0);

    // ---- GCN layer 1: 128 -> 128 ----
    gemm_wmma_kernel<128, 128, 128><<<gemmBlocks, TB, 0, stream>>>(Xb, W1p, nullptr, H, N);
    aggregate(H, AGG, 128, 7);
    bn_act(AGG, c1b, 1, Xb, 128, 7, 1);     // +bias, BN2, ReLU

    // ---- GCN layer 2: 128 -> 128 ----
    gemm_wmma_kernel<128, 128, 128><<<gemmBlocks, TB, 0, stream>>>(Xb, W2p, nullptr, H, N);
    aggregate(H, AGG, 128, 7);
    bn_act(AGG, c2b, 2, Xb, 128, 7, 1);     // +bias, BN3, ReLU

    // ---- GCN layer 3: 128 -> 64 ----
    gemm_wmma_kernel<128, 64, 64><<<gemmBlocks, TB, 0, stream>>>(Xb, W3p, nullptr, H, N);
    aggregate(H, AGG, 64, 6);
    bn_act(AGG, c3b, 3, Xb, 64, 6, 1);      // +bias, BN4, ReLU

    // ---- GCN layer 4: 64 -> 32 ----
    gemm_wmma_kernel<64, 32, 32><<<gemmBlocks, TB, 0, stream>>>(Xb, W4p, nullptr, H, N);
    aggregate(H, AGG, 32, 5);
    bn_act(AGG, c4b, 4, Xb, 32, 5, 1);      // +bias, BN5, then ReLU (pre-MLP)

    // ---- MLP layer 1: 32 -> 64 (bias fused into GEMM) ----
    gemm_wmma_kernel<32, 64, 64><<<gemmBlocks, TB, 0, stream>>>(Xb, L1p, lb1, H, N);
    bn_act(H, nullptr, 5, Xb, 64, 6, 1);    // BN6, ReLU

    // ---- MLP layer 2: 64 -> 40, write d_out [N, 40] ----
    gemm_wmma_kernel<64, 40, 48><<<gemmBlocks, TB, 0, stream>>>(Xb, L2p, lb2, (float*)d_out, N);
}
